// RecurrentAutoEncoder_71708773974241
// MI455X (gfx1250) — matrix operand order repaired and checked by reading, not compiled, not run on hardware
//
#include <hip/hip_runtime.h>
#include <hip/hip_fp16.h>

// ---------------------------------------------------------------------------
// RecurrentAutoEncoder on MI455X (gfx1250): persistent-LSTM with f16 WMMA.
// B=128, T=512, I=128, H=512, gates 4H=2048, gate order i,f,g,o.
// ---------------------------------------------------------------------------

typedef __attribute__((ext_vector_type(16))) _Float16 v16h;
typedef __attribute__((ext_vector_type(8)))  _Float16 v8h;
typedef __attribute__((ext_vector_type(8)))  float    v8f;

constexpr int Bsz  = 128;
constexpr int Tlen = 512;
constexpr int Hh   = 512;
constexpr int NWG  = 32;      // H / 16 hidden units per workgroup

#define DEVI __device__ __forceinline__

// A-matrix 16x32 f16 layout: lane half h: VGPR0-3 = K 8h..8h+7, VGPR4-7 = K 16+8h..+7
DEVI v16h loadA16(const _Float16* row, int lh) {
  v8h a = *(const v8h*)(row + 8 * lh);
  v8h b = *(const v8h*)(row + 16 + 8 * lh);
  return __builtin_shufflevector(a, b, 0, 1, 2, 3, 4, 5, 6, 7,
                                 8, 9, 10, 11, 12, 13, 14, 15);
}
// B-matrix 32x16 f16: lane = N column; half h: VGPR0-3 = K 8h..8h+7, VGPR4-7 = K 16+8h..+7
DEVI v16h loadB16(const _Float16* row, int lh) {
  v8h a = *(const v8h*)(row + 8 * lh);
  v8h b = *(const v8h*)(row + 16 + 8 * lh);
  return __builtin_shufflevector(a, b, 0, 1, 2, 3, 4, 5, 6, 7,
                                 8, 9, 10, 11, 12, 13, 14, 15);
}

DEVI v8f wmma16(v16h a, v16h b, v8f c) {
  return __builtin_amdgcn_wmma_f32_16x16x32_f16(false, a, false, b,
                                                (short)0, c, false, false);
}

DEVI float sigm(float x) { return 1.f / (1.f + __expf(-x)); }

// Generation-based grid-wide barrier (all NWG workgroups co-resident).
DEVI void grid_barrier(unsigned* cnt, unsigned* gen) {
  __syncthreads();
  if (threadIdx.x == 0) {
    __threadfence();
    unsigned g = __hip_atomic_load(gen, __ATOMIC_ACQUIRE, __HIP_MEMORY_SCOPE_AGENT);
    unsigned a = __hip_atomic_fetch_add(cnt, 1u, __ATOMIC_ACQ_REL, __HIP_MEMORY_SCOPE_AGENT);
    if (a == (unsigned)(NWG - 1)) {
      __hip_atomic_store(cnt, 0u, __ATOMIC_RELAXED, __HIP_MEMORY_SCOPE_AGENT);
      __hip_atomic_fetch_add(gen, 1u, __ATOMIC_RELEASE, __HIP_MEMORY_SCOPE_AGENT);
    } else {
      while (__hip_atomic_load(gen, __ATOMIC_ACQUIRE, __HIP_MEMORY_SCOPE_AGENT) == g) {
        __builtin_amdgcn_s_sleep(8);
      }
    }
    __threadfence();
  }
  __syncthreads();
}

// ---------------------------------------------------------------------------
// Persistent LSTM pass. Each of the 32 WGs owns 16 hidden units:
//  - LDS: Ws[64][Ktot] f16 weight slice (cols: i0..15,f0..15,g0..15,o0..15),
//         gs[128][64] f32 gate scratch, cS[128][16] f32 cell state, bias[64].
//  - Per step: gates = [x_t | h_{t-1}] @ Ws   (f16 WMMA, f32 accum),
//    then elementwise LSTM update; h double-buffered in global f16.
// ---------------------------------------------------------------------------
__global__ __launch_bounds__(256) void lstm_pass_kernel(
    const _Float16* __restrict__ Xsrc,  // [B][T][KX] f16
    int KX, int shift,                  // shift=1 => teacher forcing (read t-1)
    const _Float16* __restrict__ Wih,   // [4H][KX] f16
    const _Float16* __restrict__ Whh,   // [4H][H]  f16
    const float* __restrict__ bias,     // [4H] combined bih+bhh
    const float* __restrict__ c0,       // [B][H] f32 or null -> zero
    const _Float16* __restrict__ h0,    // [B][H] f16 or null -> zero
    _Float16* __restrict__ hbuf,        // [2][B][H] f16 working state
    _Float16* __restrict__ seqOut,      // [B][T][H] f16 or null
    _Float16* __restrict__ hsave,       // [B][H] f16 final hidden or null
    unsigned* barCnt, unsigned* barGen)
{
  const int Ktot = KX + Hh;
  extern __shared__ char smem[];
  _Float16* Ws    = (_Float16*)smem;                                       // 64*Ktot
  float*    gs    = (float*)(smem + (size_t)64 * Ktot * sizeof(_Float16)); // 128*64
  float*    cS    = gs + 128 * 64;                                         // 128*16
  float*    biasS = cS + 128 * 16;                                         // 64

  const int tid  = threadIdx.x;
  const int lane = tid & 31, wid = tid >> 5;
  const int lr = lane & 15, lh = lane >> 4;
  const int wm = wid & 3,  wn = wid >> 2;
  const int m0 = wm * 32,  n0 = wn * 32;
  const int u0 = blockIdx.x * 16;

  // --- load weight slice into LDS: Ws[n][k] = W[gate*H + u0 + n%16][k] ---
  const int kch = Ktot >> 3;          // 8-halfs (16B) chunks per row
  for (int ci = tid; ci < 64 * kch; ci += 256) {
    int n  = ci / kch;
    int kc = (ci - n * kch) << 3;
    int gate = n >> 4, ul = n & 15;
    int r = gate * Hh + u0 + ul;
    const _Float16* s = (kc < KX) ? (Wih + (size_t)r * KX + kc)
                                  : (Whh + (size_t)r * Hh + (kc - KX));
    *(v8h*)(Ws + (size_t)n * Ktot + kc) = *(const v8h*)s;
  }
  if (tid < 64) biasS[tid] = bias[(tid >> 4) * Hh + u0 + (tid & 15)];
  for (int e = tid; e < 2048; e += 256) {
    int b = e >> 4, jl = e & 15;
    cS[e] = c0 ? c0[(size_t)b * Hh + u0 + jl] : 0.f;
    hbuf[(size_t)b * Hh + u0 + jl] = h0 ? h0[(size_t)b * Hh + u0 + jl]
                                        : (_Float16)0.f;
  }
  grid_barrier(barCnt, barGen);       // full h0 visible everywhere

  const _Float16* wrow0 = Ws + (size_t)(n0 + lr) * Ktot;        // B col n0+lr
  const _Float16* wrow1 = wrow0 + (size_t)16 * Ktot;            // B col n0+16+lr

  for (int t = 0; t < Tlen; ++t) {
    const _Float16* hR = hbuf + (size_t)(t & 1) * (Bsz * Hh);
    _Float16*       hW = hbuf + (size_t)((t + 1) & 1) * (Bsz * Hh);
    const int tt = t - shift;

    v8f a00 = {}, a01 = {}, a10 = {}, a11 = {};

    // ---- input-projection part of concat-K (skipped for zero first input) ----
    if (tt >= 0) {
      const _Float16* xr0 = Xsrc + ((size_t)(m0 + lr) * Tlen + tt) * KX;
      const _Float16* xr1 = xr0 + (size_t)16 * Tlen * KX;
      for (int k0 = 0; k0 < KX; k0 += 32) {
        v16h fa0 = loadA16(xr0 + k0, lh);
        v16h fa1 = loadA16(xr1 + k0, lh);
        v16h fb0 = loadB16(wrow0 + k0, lh);
        v16h fb1 = loadB16(wrow1 + k0, lh);
        a00 = wmma16(fa0, fb0, a00);
        a01 = wmma16(fa0, fb1, a01);
        a10 = wmma16(fa1, fb0, a10);
        a11 = wmma16(fa1, fb1, a11);
      }
    }
    // ---- recurrent part: h_{t-1} ----
    {
      const _Float16* hr0 = hR + (size_t)(m0 + lr) * Hh;
      const _Float16* hr1 = hr0 + 16 * Hh;
      const _Float16* w0  = wrow0 + KX;
      const _Float16* w1  = wrow1 + KX;
      for (int k0 = 0; k0 < Hh; k0 += 32) {
        v16h fa0 = loadA16(hr0 + k0, lh);
        v16h fa1 = loadA16(hr1 + k0, lh);
        v16h fb0 = loadB16(w0 + k0, lh);
        v16h fb1 = loadB16(w1 + k0, lh);
        a00 = wmma16(fa0, fb0, a00);
        a01 = wmma16(fa0, fb1, a01);
        a10 = wmma16(fa1, fb0, a10);
        a11 = wmma16(fa1, fb1, a11);
      }
    }

    // C/D layout: lane n = lr; VGPR r -> m = 8*lh + r
#pragma unroll
    for (int r = 0; r < 8; ++r) {
      int mA = m0 + 8 * lh + r, mB = mA + 16;
      gs[mA * 64 + n0 + lr]      = a00[r];
      gs[mA * 64 + n0 + 16 + lr] = a01[r];
      gs[mB * 64 + n0 + lr]      = a10[r];
      gs[mB * 64 + n0 + 16 + lr] = a11[r];
    }
    __syncthreads();

    // elementwise LSTM update for the 128x16 (b, unit) slice
    for (int e = tid; e < 2048; e += 256) {
      int b = e >> 4, jl = e & 15;
      const float* g = gs + b * 64;
      float i  = sigm(g[jl]        + biasS[jl]);
      float f  = sigm(g[16 + jl]   + biasS[16 + jl]);
      float gg = tanhf(g[32 + jl]  + biasS[32 + jl]);
      float o  = sigm(g[48 + jl]   + biasS[48 + jl]);
      float c  = f * cS[e] + i * gg;
      cS[e] = c;
      float h = o * tanhf(c);
      _Float16 hx = (_Float16)h;
      hW[(size_t)b * Hh + u0 + jl] = hx;
      if (seqOut) seqOut[((size_t)b * Tlen + t) * Hh + u0 + jl] = hx;
      if (hsave && t == Tlen - 1) hsave[(size_t)b * Hh + u0 + jl] = hx;
    }
    grid_barrier(barCnt, barGen);     // publish h_t before next step reads it
  }
}

// ---------------------------------------------------------------------------
// FC head: out[BT,128] = d1[BT,512] @ fcW[128,512]^T + fcb   (f16 WMMA)
// ---------------------------------------------------------------------------
__global__ __launch_bounds__(256) void fc_head_kernel(
    const _Float16* __restrict__ A,
    const _Float16* __restrict__ W,
    const float* __restrict__ bias,
    float* __restrict__ out) {
  const int tid = threadIdx.x, lane = tid & 31, wid = tid >> 5;
  const int lr = lane & 15, lh = lane >> 4;
  const int wm = wid & 3, wn = wid >> 2;
  const size_t m0 = (size_t)blockIdx.x * 128 + (size_t)wm * 32;
  const int n0 = wn * 64;
  v8f acc[2][4] = {};
  const _Float16* ar0 = A + (m0 + lr) * 512;
  const _Float16* ar1 = ar0 + 16 * 512;
  for (int k0 = 0; k0 < 512; k0 += 32) {
    v16h fa0 = loadA16(ar0 + k0, lh);
    v16h fa1 = loadA16(ar1 + k0, lh);
#pragma unroll
    for (int nt = 0; nt < 4; ++nt) {
      v16h fb = loadB16(W + (size_t)(n0 + nt * 16 + lr) * 512 + k0, lh);
      acc[0][nt] = wmma16(fa0, fb, acc[0][nt]);
      acc[1][nt] = wmma16(fa1, fb, acc[1][nt]);
    }
  }
#pragma unroll
  for (int mt = 0; mt < 2; ++mt)
#pragma unroll
    for (int nt = 0; nt < 4; ++nt)
#pragma unroll
      for (int r = 0; r < 8; ++r) {
        size_t m = m0 + mt * 16 + 8 * lh + r;
        int n = n0 + nt * 16 + lr;
        out[m * 128 + n] = acc[mt][nt][r] + bias[n];
      }
}

// ---------------------------------------------------------------------------
// Prep kernels
// ---------------------------------------------------------------------------
__global__ void cvt_f16_kernel(const float* __restrict__ s,
                               _Float16* __restrict__ d, size_t n) {
  size_t i  = (size_t)blockIdx.x * blockDim.x + threadIdx.x;
  size_t st = (size_t)gridDim.x * blockDim.x;
  for (; i < n; i += st) d[i] = (_Float16)s[i];
}

__global__ void bias_comb_kernel(const float* __restrict__ a,
                                 const float* __restrict__ b,
                                 float* __restrict__ d, int n) {
  int i = blockIdx.x * blockDim.x + threadIdx.x;
  if (i < n) d[i] = a[i] + b[i];
}

__global__ void init_barrier_kernel(unsigned* p) {
  if (threadIdx.x == 0) { p[0] = 0u; p[1] = 0u; }
}

// ---------------------------------------------------------------------------
// Host launcher
// ---------------------------------------------------------------------------
extern "C" void kernel_launch(void* const* d_in, const int* in_sizes, int n_in,
                              void* d_out, int out_size, void* d_ws, size_t ws_size,
                              hipStream_t stream) {
  (void)in_sizes; (void)n_in; (void)out_size; (void)ws_size;

  const float* x     = (const float*)d_in[0];
  const float* c_dec = (const float*)d_in[1];
  const float* Wsrc[9] = {
      (const float*)d_in[2],  (const float*)d_in[3],   // enc_Wih0, enc_Whh0
      (const float*)d_in[6],  (const float*)d_in[7],   // enc_Wih1, enc_Whh1
      (const float*)d_in[10], (const float*)d_in[11],  // dec_Wih0, dec_Whh0
      (const float*)d_in[14], (const float*)d_in[15],  // dec_Wih1, dec_Whh1
      (const float*)d_in[18]                           // fc_W
  };
  const size_t Wn[9] = { 2048u*128, 2048u*512, 2048u*512, 2048u*512,
                         2048u*128, 2048u*512, 2048u*512, 2048u*512, 128u*512 };
  const float* bih[4] = { (const float*)d_in[4],  (const float*)d_in[8],
                          (const float*)d_in[12], (const float*)d_in[16] };
  const float* bhh[4] = { (const float*)d_in[5],  (const float*)d_in[9],
                          (const float*)d_in[13], (const float*)d_in[17] };
  const float* fcb = (const float*)d_in[19];

  // bump-allocate workspace (256B aligned)
  char* p = (char*)d_ws;
  auto take = [&](size_t bytes) -> char* {
    char* r = p; p += (bytes + 255) & ~(size_t)255; return r;
  };
  _Float16* xb   = (_Float16*)take((size_t)Bsz * Tlen * 128 * 2);
  _Float16* seqA = (_Float16*)take((size_t)Bsz * Tlen * Hh * 2);
  _Float16* seqB = (_Float16*)take((size_t)Bsz * Tlen * Hh * 2);
  _Float16* hbuf = (_Float16*)take((size_t)2 * Bsz * Hh * 2);
  _Float16* hs0  = (_Float16*)take((size_t)Bsz * Hh * 2);
  _Float16* hs1  = (_Float16*)take((size_t)Bsz * Hh * 2);
  _Float16* wf[9];
  for (int i = 0; i < 9; ++i) wf[i] = (_Float16*)take(Wn[i] * 2);
  float* biasc[4];
  for (int i = 0; i < 4; ++i) biasc[i] = (float*)take(2048 * 4);
  unsigned* bar = (unsigned*)take(256);

  // --- prep: f32 -> f16 conversions, combined biases, barrier init ---
  cvt_f16_kernel<<<1024, 256, 0, stream>>>(x, xb, (size_t)Bsz * Tlen * 128);
  for (int i = 0; i < 9; ++i)
    cvt_f16_kernel<<<512, 256, 0, stream>>>(Wsrc[i], wf[i], Wn[i]);
  for (int i = 0; i < 4; ++i)
    bias_comb_kernel<<<8, 256, 0, stream>>>(bih[i], bhh[i], biasc[i], 2048);
  init_barrier_kernel<<<1, 32, 0, stream>>>(bar);

  const size_t elemBytes = (size_t)(128 * 64 + 128 * 16 + 64) * 4;
  const size_t sm0 = (size_t)64 * (128 + Hh) * 2 + elemBytes;  // Ktot=640
  const size_t sm1 = (size_t)64 * (Hh + Hh) * 2 + elemBytes;   // Ktot=1024
  (void)hipFuncSetAttribute(reinterpret_cast<const void*>(lstm_pass_kernel),
                            hipFuncAttributeMaxDynamicSharedMemorySize, (int)sm1);

  // encoder layer 0: x -> seqA (e0), final hidden -> hs0
  lstm_pass_kernel<<<NWG, 256, sm0, stream>>>(xb, 128, 0, wf[0], wf[1], biasc[0],
      nullptr, nullptr, hbuf, seqA, hs0, bar, bar + 1);
  // encoder layer 1: e0 -> (discard seq), final hidden -> hs1
  lstm_pass_kernel<<<NWG, 256, sm1, stream>>>(seqA, 512, 0, wf[2], wf[3], biasc[1],
      nullptr, nullptr, hbuf, nullptr, hs1, bar, bar + 1);
  // decoder layer 0: shifted x, h0=hs0, c0=c_dec[0] -> seqA (d0)
  lstm_pass_kernel<<<NWG, 256, sm0, stream>>>(xb, 128, 1, wf[4], wf[5], biasc[2],
      c_dec, hs0, hbuf, seqA, nullptr, bar, bar + 1);
  // decoder layer 1: d0, h0=hs1, c0=c_dec[1] -> seqB (d1)
  lstm_pass_kernel<<<NWG, 256, sm1, stream>>>(seqA, 512, 0, wf[6], wf[7], biasc[3],
      c_dec + (size_t)Bsz * Hh, hs1, hbuf, seqB, nullptr, bar, bar + 1);

  // FC head: d1 @ fc_W^T + fc_b  -> d_out [B,T,128] f32
  fc_head_kernel<<<(Bsz * Tlen) / 128, 256, 0, stream>>>(seqB, wf[8], fcb,
                                                         (float*)d_out);
}
